// seqFusionAttentionWeightProjCP_48782238548115
// MI455X (gfx1250) — matrix operand order repaired and checked by reading, not compiled, not run on hardware
//
#include <hip/hip_runtime.h>

typedef __attribute__((ext_vector_type(16))) __bf16 v16bf;
typedef __attribute__((ext_vector_type(8)))  float  v8f;

#define BB   4
#define SS   2048
#define HH   4
#define CC   64
#define DD   256
#define WINW 10

#define WBLK (64 * 264)   // one padded, transposed 64-col weight block (bf16 elems)

static __device__ __forceinline__ v8f wmma_bf16(v16bf a, v16bf b, v8f c) {
  // D = A(16x32 bf16) * B(32x16 bf16) + C(16x16 f32)
  return __builtin_amdgcn_wmma_f32_16x16x32_bf16(false, a, false, b, (short)0, c,
                                                 false, false);
}

// Async DMA of one 16B chunk from global into LDS (gfx1250 ASYNCcnt path).
static __device__ __forceinline__ void async_g2l_b128(void* lds_dst,
                                                      const void* gsrc) {
  unsigned la = (unsigned)(uintptr_t)lds_dst;          // low 32 bits = LDS addr
  unsigned long long ga = (unsigned long long)(uintptr_t)gsrc;
  asm volatile("global_load_async_to_lds_b128 %0, %1, off"
               :: "v"(la), "v"(ga) : "memory");
}
static __device__ __forceinline__ void wait_async0() {
  asm volatile("s_wait_asynccnt 0x0" ::: "memory");
}
static __device__ __forceinline__ void wait_ds0() {
  asm volatile("s_wait_dscnt 0x0" ::: "memory");
}

// ---------------------------------------------------------------------------
// Kernel 0: weight prep. Convert all 5 fp32 weight matrices to bf16, stored as
// padded, transposed 64-column blocks: WtAll[blk][n(64)][k(264)], blk = z*4+h
// for Q/K/V/G (z=0..3) and blk = 16+nt for OW. This is exactly the LDS image
// the GEMM kernels want, so they can DMA it in with one contiguous async copy.
// ---------------------------------------------------------------------------
__global__ __launch_bounds__(256)
void cp_wprep_kernel(const float* __restrict__ Wq, const float* __restrict__ Wk,
                     const float* __restrict__ Wv, const float* __restrict__ Wg,
                     const float* __restrict__ OW, __bf16* __restrict__ WtAll) {
  const int zz = blockIdx.x;  // 0..19
  const float* W;
  int col0;
  if (zz < 16) {
    W = (zz < 4) ? Wq : (zz < 8) ? Wk : (zz < 12) ? Wv : Wg;
    col0 = (zz & 3) * 64;
  } else {
    W = OW;
    col0 = (zz - 16) * 64;
  }
  __bf16* dst = WtAll + (size_t)zz * WBLK;
  // coalesced reads over n; k-major per iteration
  for (int i = threadIdx.x; i < 64 * 264; i += 256) {
    int k = i >> 6;        // 0..263  (i = k*64 + n)
    int n = i & 63;
    float v = (k < 256) ? W[k * DD + col0 + n] : 0.f;
    dst[n * 264 + k] = (__bf16)v;
  }
}

// ---------------------------------------------------------------------------
// Kernel 1: fused Q/K/V/G projections.
//   X[8192,256](f32) @ W[256,256] -> [B][H][S][C] bf16 (G: sigmoid(x+GB))
//   grid = (128 Mtiles, 4 heads, 4 matrices), block = 128 (4 waves)
// ---------------------------------------------------------------------------
__global__ __launch_bounds__(128)
void cp_proj_kernel(const float* __restrict__ Qin, const float* __restrict__ Kin,
                    const float* __restrict__ Vin, const float* __restrict__ GB,
                    const __bf16* __restrict__ WtAll,
                    __bf16* __restrict__ Qbf, __bf16* __restrict__ Kbf,
                    __bf16* __restrict__ Vbf, __bf16* __restrict__ Gbf) {
  const int mt = blockIdx.x, h = blockIdx.y, z = blockIdx.z;
  const float* X = (z == 0) ? Qin : ((z == 1) ? Kin : Vin);  // G uses Vin
  __bf16* Out    = (z == 0) ? Qbf : ((z == 1) ? Kbf : ((z == 2) ? Vbf : Gbf));

  __shared__ __bf16 sBT[WBLK];  // [n][k], stride 264 (conflict-free frag reads)

  const int lane = threadIdx.x & 31, wave = threadIdx.x >> 5;
  const int mlo = lane & 15, hi = lane >> 4;
  const int offA = hi ? 16 : 0;   // contiguous k order used here: lanes16-31 hold K 16..31
  const int offB = hi ? 16 : 0;   // contiguous k order used here: lanes16-31 hold K 16..31
  const int mbase = mt * 64 + wave * 16;
  const float* xrow = X + (size_t)(mbase + mlo) * DD;

  // warm A rows in L2/L0 while the weight DMA runs
  __builtin_prefetch(xrow, 0, 0);
  __builtin_prefetch(xrow + 128, 0, 0);

  // async-DMA the pre-transposed bf16 weight block into LDS (33792 B)
  const __bf16* wsrc = WtAll + (size_t)(z * 4 + h) * WBLK;
  for (int ch = threadIdx.x; ch < WBLK / 8; ch += 128)
    async_g2l_b128(&sBT[ch * 8], wsrc + ch * 8);
  wait_async0();
  __syncthreads();

  v8f zero = {0.f, 0.f, 0.f, 0.f, 0.f, 0.f, 0.f, 0.f};
  v8f acc[4] = {zero, zero, zero, zero};

#pragma unroll
  for (int kc = 0; kc < DD; kc += 32) {
    v16bf a;
#pragma unroll
    for (int e = 0; e < 8; ++e) {
      a[e]     = (__bf16)xrow[kc + offA + e];
      a[e + 8] = (__bf16)xrow[kc + 8 + offA + e];
    }
#pragma unroll
    for (int nt = 0; nt < 4; ++nt) {
      const __bf16* col = &sBT[(nt * 16 + mlo) * 264 + kc + offB];
      v16bf bm;
#pragma unroll
      for (int e = 0; e < 16; ++e) bm[e] = col[e];
      acc[nt] = wmma_bf16(a, bm, acc[nt]);
    }
  }

#pragma unroll
  for (int nt = 0; nt < 4; ++nt) {
    int c = nt * 16 + mlo;
    float gb = (z == 3) ? GB[h * 64 + c] : 0.f;
#pragma unroll
    for (int r = 0; r < 8; ++r) {
      int grow = mbase + r + hi * 8;           // C-layout: m = r + 8*(lane/16)
      int b = grow >> 11, s = grow & (SS - 1);
      float v = acc[nt][r];
      if (z == 3) v = __builtin_amdgcn_rcpf(1.f + __expf(-(v + gb)));  // sigmoid
      Out[((size_t)(b * HH + h) * SS + s) * CC + c] = (__bf16)v;
    }
  }
}

// ---------------------------------------------------------------------------
// Kernel 2: banded attention. One wave per (b, h, 16-query tile).
//   scores(3x WMMA pairs) -> band mask -> fp32 softmax -> P@V (WMMA) -> *G
//   grid = 512 blocks x 128 threads (4 independent waves/block)
// ---------------------------------------------------------------------------
__global__ __launch_bounds__(128)
void cp_attn_kernel(const __bf16* __restrict__ Qbf, const __bf16* __restrict__ Kbf,
                    const __bf16* __restrict__ Vbf, const __bf16* __restrict__ Gbf,
                    const unsigned char* __restrict__ seqMask,
                    __bf16* __restrict__ Obuf) {
  __shared__ float  scoreS[4][16 * 48];
  __shared__ __bf16 pbufS[4][16 * 72];   // P in A-operand rows, padded stride
  __shared__ __bf16 vtS[4][64 * 50];     // V transposed: vt[c][t], odd dword stride

  const int wave = threadIdx.x >> 5, lane = threadIdx.x & 31;
  const int task = blockIdx.x * 4 + wave;           // 0..2047
  const int qt = task & 127, h = (task >> 7) & 3, b = task >> 9;
  const int s0 = qt * 16;

  float*  score = scoreS[wave];
  __bf16* pbuf  = pbufS[wave];
  __bf16* vt    = vtS[wave];

  const size_t hb = (size_t)(b * HH + h) * SS;
  const __bf16* Qh = Qbf + hb * CC;
  const __bf16* Kh = Kbf + hb * CC;
  const __bf16* Vh = Vbf + hb * CC;
  const __bf16* Gh = Gbf + hb * CC;

  const int mlo = lane & 15, hi = lane >> 4;
  const int offA = hi ? 16 : 0, offB = hi ? 16 : 0;

  __builtin_prefetch(Gh + (size_t)s0 * CC, 0, 0);   // G is consumed last

  // Q fragments (K-dim = C = 64 -> two 16x32 A fragments)
  const __bf16* qrow = Qh + (size_t)(s0 + mlo) * CC;
  v16bf aq0, aq1;
#pragma unroll
  for (int e = 0; e < 8; ++e) {
    aq0[e] = qrow[offA + e];       aq0[e + 8] = qrow[8 + offA + e];
    aq1[e] = qrow[32 + offA + e];  aq1[e + 8] = qrow[40 + offA + e];
  }

  v8f zero = {0.f, 0.f, 0.f, 0.f, 0.f, 0.f, 0.f, 0.f};

  // Scores for the 3 key tiles covering the +/-10 band
#pragma unroll
  for (int j = 0; j < 3; ++j) {
    const int t0 = s0 + (j - 1) * 16;
    if (t0 >= 0 && t0 < SS) {
      const __bf16* krow = Kh + (size_t)(t0 + mlo) * CC;
      v16bf bk0, bk1;
#pragma unroll
      for (int e = 0; e < 16; ++e) {
        bk0[e] = krow[offB + e];          // B operand: lane n = key, elems = c
        bk1[e] = krow[32 + offB + e];
      }
      v8f sc = zero;
      sc = wmma_bf16(aq0, bk0, sc);
      sc = wmma_bf16(aq1, bk1, sc);
      const int t = t0 + mlo;
      const bool tmask = seqMask[b * SS + t] != 0;
#pragma unroll
      for (int r = 0; r < 8; ++r) {
        int m = r + hi * 8, s = s0 + m;
        int d = s - t;
        bool valid = (d <= WINW) && (d >= -WINW) && !tmask;
        score[m * 48 + j * 16 + mlo] = valid ? sc[r] * 0.125f : -3.0e38f;
      }
    } else {
#pragma unroll
      for (int r = 0; r < 8; ++r)
        score[(r + hi * 8) * 48 + j * 16 + mlo] = -3.0e38f;
    }
  }
  wait_ds0();   // in-wave LDS RAW fence

  // fp32 softmax over the 48 candidate keys; lanes 0..15 each own one row
  if (lane < 16) {
    const int m = lane;
    float mx = -3.0e38f;
    for (int c = 0; c < 48; ++c) mx = fmaxf(mx, score[m * 48 + c]);
    float sum = 0.f;
    for (int c = 0; c < 48; ++c) {
      float e = __expf(score[m * 48 + c] - mx);
      score[m * 48 + c] = e;
      sum += e;
    }
    float inv = __builtin_amdgcn_rcpf(sum);
    for (int c = 0; c < 48; ++c) pbuf[m * 72 + c] = (__bf16)(score[m * 48 + c] * inv);
    for (int c = 48; c < 64; ++c) pbuf[m * 72 + c] = (__bf16)0.f;  // zero pad K
  }

  // Stage V transposed: vt[c][t], t in [s0-16, s0+32)
  for (int idx = lane; idx < 48 * 64; idx += 32) {
    int tt = idx >> 6, c = idx & 63;
    int gt = s0 - 16 + tt;
    __bf16 v = (gt >= 0 && gt < SS) ? Vh[(size_t)gt * CC + c] : (__bf16)0.f;
    vt[c * 50 + tt] = v;
  }
  wait_ds0();

  // out(16x64) = P(16x64) @ Vt  via 2 K-steps x 4 C-tiles
  v8f ao[4] = {zero, zero, zero, zero};
#pragma unroll
  for (int kc = 0; kc < 64; kc += 32) {
    const __bf16* prow = pbuf + mlo * 72 + kc;
    v16bf pa;
#pragma unroll
    for (int e = 0; e < 8; ++e) {
      pa[e]     = prow[offA + e];
      pa[e + 8] = prow[8 + offA + e];
    }
#pragma unroll
    for (int cn = 0; cn < 4; ++cn) {
      const __bf16* vcol = vt + (cn * 16 + mlo) * 50 + kc + offB;
      v16bf bv;
#pragma unroll
      for (int e = 0; e < 16; ++e) bv[e] = vcol[e];
      ao[cn] = wmma_bf16(pa, bv, ao[cn]);
    }
  }

  // gate with G and emit bf16 pre-projection activations [B][S][H*C]
#pragma unroll
  for (int cn = 0; cn < 4; ++cn) {
    int c = cn * 16 + mlo;
#pragma unroll
    for (int r = 0; r < 8; ++r) {
      int m = r + hi * 8, s = s0 + m;
      float g = (float)Gh[(size_t)s * CC + c];
      float val = ao[cn][r] * g;
      Obuf[((size_t)(b * SS + s)) * DD + h * CC + c] = (__bf16)val;
    }
  }
}

// ---------------------------------------------------------------------------
// Kernel 3: output projection  Obuf[8192,256](bf16) @ OW + OB -> f32 out
//   grid = (128 Mtiles, 4 Ntiles), block = 128 (4 waves)
// ---------------------------------------------------------------------------
__global__ __launch_bounds__(128)
void cp_outproj_kernel(const __bf16* __restrict__ Obuf,
                       const __bf16* __restrict__ WtAll,
                       const float* __restrict__ OB, float* __restrict__ out) {
  const int mt = blockIdx.x, n0 = blockIdx.y * 64;

  __shared__ __bf16 sBT[WBLK];

  const int lane = threadIdx.x & 31, wave = threadIdx.x >> 5;
  const int mlo = lane & 15, hi = lane >> 4;
  const int offA = hi ? 16 : 0, offB = hi ? 16 : 0;
  const int mbase = mt * 64 + wave * 16;
  const __bf16* arow = Obuf + (size_t)(mbase + mlo) * DD;

  __builtin_prefetch(arow, 0, 0);

  const __bf16* wsrc = WtAll + (size_t)(16 + blockIdx.y) * WBLK;
  for (int ch = threadIdx.x; ch < WBLK / 8; ch += 128)
    async_g2l_b128(&sBT[ch * 8], wsrc + ch * 8);
  wait_async0();
  __syncthreads();

  v8f zero = {0.f, 0.f, 0.f, 0.f, 0.f, 0.f, 0.f, 0.f};
  v8f acc[4] = {zero, zero, zero, zero};

#pragma unroll
  for (int kc = 0; kc < DD; kc += 32) {
    v16bf a;
#pragma unroll
    for (int e = 0; e < 8; ++e) {
      a[e]     = arow[kc + offA + e];
      a[e + 8] = arow[kc + 8 + offA + e];
    }
#pragma unroll
    for (int nt = 0; nt < 4; ++nt) {
      const __bf16* col = &sBT[(nt * 16 + mlo) * 264 + kc + offB];
      v16bf bm;
#pragma unroll
      for (int e = 0; e < 16; ++e) bm[e] = col[e];
      acc[nt] = wmma_bf16(a, bm, acc[nt]);
    }
  }

#pragma unroll
  for (int nt = 0; nt < 4; ++nt) {
    int n = n0 + nt * 16 + mlo;
    float bias = OB[n];
#pragma unroll
    for (int r = 0; r < 8; ++r) {
      int grow = mbase + r + hi * 8;
      out[(size_t)grow * DD + n] = acc[nt][r] + bias;
    }
  }
}

// ---------------------------------------------------------------------------
extern "C" void kernel_launch(void* const* d_in, const int* in_sizes, int n_in,
                              void* d_out, int out_size, void* d_ws, size_t ws_size,
                              hipStream_t stream) {
  (void)in_sizes; (void)n_in; (void)out_size; (void)ws_size;
  const float* Qin = (const float*)d_in[0];
  const float* Kin = (const float*)d_in[1];
  const float* Vin = (const float*)d_in[2];
  const float* Wq  = (const float*)d_in[3];
  const float* Wk  = (const float*)d_in[4];
  const float* Wv  = (const float*)d_in[5];
  const float* Wg  = (const float*)d_in[6];
  const float* GB  = (const float*)d_in[7];
  const float* OW  = (const float*)d_in[8];
  const float* OB  = (const float*)d_in[9];
  const unsigned char* seqMask = (const unsigned char*)d_in[10];

  char* ws = (char*)d_ws;
  const size_t SZ = (size_t)BB * HH * SS * CC * sizeof(__bf16);  // 4 MB each
  __bf16* Qbf   = (__bf16*)(ws);
  __bf16* Kbf   = (__bf16*)(ws + SZ);
  __bf16* Vbf   = (__bf16*)(ws + 2 * SZ);
  __bf16* Gbf   = (__bf16*)(ws + 3 * SZ);
  __bf16* Obuf  = (__bf16*)(ws + 4 * SZ);           // [B][S][H*C] bf16, 4 MB
  __bf16* WtAll = (__bf16*)(ws + 5 * SZ);           // 20 blocks * 33792 B
  float* out = (float*)d_out;

  cp_wprep_kernel<<<20, 256, 0, stream>>>(Wq, Wk, Wv, Wg, OW, WtAll);

  dim3 g1(128, 4, 4);
  cp_proj_kernel<<<g1, 128, 0, stream>>>(Qin, Kin, Vin, GB, WtAll,
                                         Qbf, Kbf, Vbf, Gbf);

  cp_attn_kernel<<<512, 128, 0, stream>>>(Qbf, Kbf, Vbf, Gbf, seqMask, Obuf);

  dim3 g3(128, 4);
  cp_outproj_kernel<<<g3, 128, 0, stream>>>(Obuf, WtAll, OB, out);
}
